// SelfAttention_4836133175620
// MI455X (gfx1250) — matrix operand order repaired and checked by reading, not compiled, not run on hardware
//
#include <hip/hip_runtime.h>
#include <hip/hip_bf16.h>
#include <math.h>

#define DEVINL __device__ __forceinline__

typedef _Float16 h16;
typedef __attribute__((ext_vector_type(16))) _Float16 v16h;
typedef __attribute__((ext_vector_type(8)))  _Float16 v8h;
typedef __attribute__((ext_vector_type(8)))  float    v8f;
typedef int v4i __attribute__((vector_size(16)));

constexpr int Bb = 4;
constexpr int Cc = 1024;
constexpr int Tt = 2048;
constexpr int Hh = 16;
constexpr int DH = 64;

#if defined(__has_builtin)
#if __has_builtin(__builtin_amdgcn_global_load_async_to_lds_b128) && \
    __has_builtin(__builtin_amdgcn_s_wait_asynccnt)
#define HAS_ASYNC_LDS 1
#endif
#endif

DEVINL v8f wmma16x16x32(v16h a, v16h b, v8f c) {
  return __builtin_amdgcn_wmma_f32_16x16x32_f16(
      false, a, false, b, (short)0, c, false, false);
}

// A fragment (16x32, M x K) from ROW-MAJOR tile: per lane two contiguous 16B
// runs (k in [8hi,8hi+8) and [16+8hi,24+8hi)) -> 2x ds_load_b128.
DEVINL v16h load_frag_a_rm(const h16* p, int ld) {
  const int lane = threadIdx.x & 31;
  const int m = lane & 15, hi = lane >> 4;
  const h16* row = p + m * ld + hi * 8;
  v8h lo = *(const v8h*)(row);
  v8h hh = *(const v8h*)(row + 16);
  return __builtin_shufflevector(lo, hh, 0, 1, 2, 3, 4, 5, 6, 7,
                                 8, 9, 10, 11, 12, 13, 14, 15);
}

// B fragment (32x16, K x N) from N-MAJOR tile [n][k]: per lane two contiguous
// 16B runs (k in [8hi,8hi+8) and [16+8hi,24+8hi)).
DEVINL v16h load_frag_b_nm(const h16* p, int ld) {
  const int lane = threadIdx.x & 31;
  const int n = lane & 15, hi = lane >> 4;
  const h16* col = p + n * ld + hi * 8;
  v8h lo = *(const v8h*)(col);
  v8h hh = *(const v8h*)(col + 16);
  return __builtin_shufflevector(lo, hh, 0, 1, 2, 3, 4, 5, 6, 7,
                                 8, 9, 10, 11, 12, 13, 14, 15);
}

DEVINL v8h pack8(const float* v) {
  v8h r;
#pragma unroll
  for (int j = 0; j < 8; ++j) r[j] = (h16)v[j];
  return r;
}

#ifdef HAS_ASYNC_LDS
DEVINL void async_cp16(h16* lds, const h16* g) {
  __builtin_amdgcn_global_load_async_to_lds_b128(
      (__attribute__((address_space(1))) v4i*)g,
      (__attribute__((address_space(3))) v4i*)lds, 0, 0);
}
DEVINL void async_cp32(h16* lds, const h16* g) {
  async_cp16(lds, g);
  async_cp16(lds + 8, g + 8);
}
#endif

// ---------------------------------------------------------------------------
// Kernel 0a: W (f32 [o][c]) -> wh (f16 [p][o][c])
// ---------------------------------------------------------------------------
__global__ __launch_bounds__(256)
void convert_w_kernel(const float* __restrict__ Wq, const float* __restrict__ Wk,
                      const float* __restrict__ Wv, h16* __restrict__ wh) {
  const int p = blockIdx.y;
  const float* W = (p == 0) ? Wq : (p == 1) ? Wk : Wv;
  h16* dst = wh + (size_t)p * Cc * Cc;
  const size_t i = ((size_t)blockIdx.x * 256 + threadIdx.x) * 8;
  float tmp[8];
  *(float4*)(&tmp[0]) = *(const float4*)(W + i);
  *(float4*)(&tmp[4]) = *(const float4*)(W + i + 4);
  *(v8h*)(dst + i) = pack8(tmp);
}

// ---------------------------------------------------------------------------
// Kernel 0b: x (f32 [b][c][t]) -> xh_t (f16 [b][t][c])  (LDS-tiled transpose)
// ---------------------------------------------------------------------------
__global__ __launch_bounds__(256)
void transpose_x_kernel(const float* __restrict__ x, h16* __restrict__ xh_t) {
  constexpr int TD = 64, LDP = 72;  // 144B row stride (16B aligned)
  __shared__ h16 TT[TD * LDP];
  const int t0 = blockIdx.x * TD;
  const int c0 = blockIdx.y * TD;
  const int b  = blockIdx.z;
  const int tid = threadIdx.x;
  const float* xb = x + (size_t)b * Cc * Tt;
  h16* xtb = xh_t + (size_t)b * Tt * Cc;
#pragma unroll
  for (int g = 0; g < 2; ++g) {                 // read along t, packed LDS store
    const int i = g * 2048 + tid * 8;
    const int c = i >> 6, t = i & 63;
    const float* src = xb + (size_t)(c0 + c) * Tt + t0 + t;
    float tmp[8];
    *(float4*)(&tmp[0]) = *(const float4*)(src);
    *(float4*)(&tmp[4]) = *(const float4*)(src + 4);
    *(v8h*)(&TT[c * LDP + t]) = pack8(tmp);
  }
  __syncthreads();
#pragma unroll
  for (int g = 0; g < 2; ++g) {                 // gather along c, packed store
    const int i = g * 2048 + tid * 8;
    const int t = i >> 6, c = i & 63;
    v8h pk;
#pragma unroll
    for (int j = 0; j < 8; ++j) pk[j] = TT[(c + j) * LDP + t];
    *(v8h*)(&xtb[(size_t)(t0 + t) * Cc + c0 + c]) = pk;
  }
}

// ---------------------------------------------------------------------------
// Kernel 1: QKV projection from f16 inputs, double-buffered async-LDS pipeline.
// y[b,t,o] = sum_c W[o,c]*x[b,c,t].  Q,K -> [b][h][t][d]; V -> [b][h][d][t].
// Block tile 128(M=o) x 128(N=t), K-chunk 32, 8 waves (4x2), 8 WMMA / chunk.
// ---------------------------------------------------------------------------
__global__ __launch_bounds__(256)
void qkv_proj_kernel(const h16* __restrict__ xh_t,   // [b][t][c]
                     const h16* __restrict__ wh,     // [p][o][c]
                     h16* __restrict__ q_ws,
                     h16* __restrict__ k_ws,
                     h16* __restrict__ v_ws) {
  constexpr int PM = 128, PN = 128, PK = 32;
  __shared__ h16 Wtb[2][PM * PK];   // row-major [m][k] (A side)
  __shared__ h16 Xtb[2][PN * PK];   // n-major  [t][c] (B side)

  const int tid   = threadIdx.x;
  const int wave  = tid >> 5;
  const int waveM = wave & 3;
  const int waveN = wave >> 2;
  const int nBase = blockIdx.x * PN;
  const int mBase = blockIdx.y * PM;
  const int z = blockIdx.z;
  const int b = z / 3;
  const int p = z - 3 * b;
  const h16* Wp  = wh + (size_t)p * Cc * Cc;
  h16* outw      = (p == 0) ? q_ws : (p == 1) ? k_ws : v_ws;
  const h16* xbt = xh_t + (size_t)b * Tt * Cc;

  // fill personality: row = tid>>1 (0..127), halves [seg, seg+16)
  const int frow = tid >> 1;
  const int fseg = (tid & 1) * 16;

  v8f acc[2][4];
#pragma unroll
  for (int mi = 0; mi < 2; ++mi)
#pragma unroll
    for (int ni = 0; ni < 4; ++ni) {
      v8f zv = {};
      acc[mi][ni] = zv;
    }

#define PROJ_FILL(buf, k0)                                                \
  do {                                                                    \
    const h16* wsrc = Wp + (size_t)(mBase + frow) * Cc + (k0) + fseg;     \
    const h16* xsrc = xbt + (size_t)(nBase + frow) * Cc + (k0) + fseg;    \
    h16* wdst = &Wtb[buf][frow * PK + fseg];                              \
    h16* xdst = &Xtb[buf][frow * PK + fseg];                              \
    PROJ_CP(wdst, wsrc);                                                  \
    PROJ_CP(xdst, xsrc);                                                  \
  } while (0)
#ifdef HAS_ASYNC_LDS
#define PROJ_CP(d, s) async_cp32((d), (s))
#else
#define PROJ_CP(d, s) (*(v16h*)(d) = *(const v16h*)(s))
#endif

  PROJ_FILL(0, 0);  // prologue

  constexpr int NCH = Cc / PK;  // 32
  for (int it = 0; it < NCH; ++it) {
    const int cur = it & 1;
    const bool more = (it + 1) < NCH;
    if (more) PROJ_FILL(cur ^ 1, (it + 1) * PK);
#ifdef HAS_ASYNC_LDS
    if (more) __builtin_amdgcn_s_wait_asynccnt(4);  // in-order: chunk it done
    else      __builtin_amdgcn_s_wait_asynccnt(0);
#endif
    __syncthreads();

    v16h afr[2], bfr[4];
#pragma unroll
    for (int mi = 0; mi < 2; ++mi)
      afr[mi] = load_frag_a_rm(&Wtb[cur][(waveM * 32 + mi * 16) * PK], PK);
#pragma unroll
    for (int ni = 0; ni < 4; ++ni)
      bfr[ni] = load_frag_b_nm(&Xtb[cur][(waveN * 64 + ni * 16) * PK], PK);
#pragma unroll
    for (int mi = 0; mi < 2; ++mi)
#pragma unroll
      for (int ni = 0; ni < 4; ++ni)
        acc[mi][ni] = wmma16x16x32(afr[mi], bfr[ni], acc[mi][ni]);

    __syncthreads();  // everyone done with buf `cur` before it is refilled
  }
#undef PROJ_FILL
#undef PROJ_CP

  // Epilogue: D-frag rows r=0..7 -> 8 consecutive d inside one head.
  const int lane = tid & 31;
  const int n = lane & 15, hi = lane >> 4;
#pragma unroll
  for (int mi = 0; mi < 2; ++mi)
#pragma unroll
    for (int ni = 0; ni < 4; ++ni) {
      const int o0 = mBase + waveM * 32 + mi * 16 + 8 * hi;
      const int hd = o0 >> 6, d0 = o0 & 63;
      const int t  = nBase + waveN * 64 + ni * 16 + n;
      if (p < 2) {  // Q,K: [b][h][t][d] -> one b128 f16 store
        v8h pk;
#pragma unroll
        for (int r = 0; r < 8; ++r) pk[r] = (h16)acc[mi][ni][r];
        *(v8h*)(&outw[(((size_t)(b * Hh + hd)) * Tt + t) * DH + d0]) = pk;
      } else {      // V: [b][h][d][t] -> b16 stores, coalesced across lanes
        h16* vp = &outw[(((size_t)(b * Hh + hd)) * DH + d0) * Tt + t];
#pragma unroll
        for (int r = 0; r < 8; ++r) vp[(size_t)r * Tt] = (h16)acc[mi][ni][r];
      }
    }
}

// ---------------------------------------------------------------------------
// Kernel 2: flash attention, double-buffered async K/V tiles.
// Block = 128 q-rows of one (b,h); 8 waves; Q fragments in registers.
// ---------------------------------------------------------------------------
__global__ __launch_bounds__(256)
void attn_kernel(const h16* __restrict__ q_ws,
                 const h16* __restrict__ k_ws,
                 const h16* __restrict__ v_ws,
                 const float* __restrict__ mask,
                 const float* __restrict__ bias,
                 float* __restrict__ out) {
  constexpr int QB = 128;
  constexpr int KB = 64;
  __shared__ h16 Ksb[2][KB * DH];    // [key][d] (B side of Q*K^T)
  __shared__ h16 Vtb[2][DH * KB];    // [d][key] (B side of P*V)
  __shared__ h16 Ps[8 * 16 * KB];    // per-wave P tile [m][key]
  __shared__ float Msk[2][KB];

  const int tid  = threadIdx.x;
  const int wave = tid >> 5;
  const int lane = tid & 31;
  const int n = lane & 15, hi = lane >> 4;
  const int q0 = blockIdx.x * QB;
  const int bh = blockIdx.y;
  const int b = bh >> 4;
  const int h = bh & 15;

  const h16* Qg = q_ws + ((size_t)bh * Tt + q0) * DH;
  const h16* Kg = k_ws + (size_t)bh * Tt * DH;   // [t][d]
  const h16* Vg = v_ws + (size_t)bh * DH * Tt;   // [d][t]

  // Q A-fragments straight from global.
  const h16* qrow = Qg + (size_t)(wave * 16 + n) * DH;
  v16h qa0, qa1;
  {
    v8h l0 = *(const v8h*)(qrow + hi * 8);
    v8h h0 = *(const v8h*)(qrow + 16 + hi * 8);
    qa0 = __builtin_shufflevector(l0, h0, 0, 1, 2, 3, 4, 5, 6, 7,
                                  8, 9, 10, 11, 12, 13, 14, 15);
    v8h l1 = *(const v8h*)(qrow + 32 + hi * 8);
    v8h h1 = *(const v8h*)(qrow + 48 + hi * 8);
    qa1 = __builtin_shufflevector(l1, h1, 0, 1, 2, 3, 4, 5, 6, 7,
                                  8, 9, 10, 11, 12, 13, 14, 15);
  }

  const float scale = 0.125f;
  const float bb = bias[0];

  float mrun[8], lrun[8];
  v8f oacc[4];
#pragma unroll
  for (int r = 0; r < 8; ++r) { mrun[r] = -3.0e38f; lrun[r] = 0.0f; }
#pragma unroll
  for (int di = 0; di < 4; ++di) {
    v8f zv = {};
    oacc[di] = zv;
  }

  h16* Pw = &Ps[wave * 16 * KB];

  const int koff = tid * 16;        // Ks flat offset
  const int vd   = tid >> 2;        // Vt row (d)
  const int vseg = (tid & 3) * 16;  // Vt key segment

#define ATTN_FILL(buf, kt0)                                              \
  do {                                                                   \
    const h16* ksrc = Kg + (size_t)(kt0) * DH + koff;                    \
    const h16* vsrc = Vg + (size_t)vd * Tt + (kt0) + vseg;               \
    ATTN_CP(&Ksb[buf][koff], ksrc);                                      \
    ATTN_CP(&Vtb[buf][vd * KB + vseg], vsrc);                            \
    if (tid < KB) Msk[buf][tid] = mask[(size_t)b * Tt + (kt0) + tid];    \
  } while (0)
#ifdef HAS_ASYNC_LDS
#define ATTN_CP(d, s) async_cp32((d), (s))
#else
#define ATTN_CP(d, s) (*(v16h*)(d) = *(const v16h*)(s))
#endif

  ATTN_FILL(0, 0);  // prologue

  constexpr int NIT = Tt / KB;  // 32
  for (int it = 0; it < NIT; ++it) {
    const int cur = it & 1;
    const bool more = (it + 1) < NIT;
    if (more) ATTN_FILL(cur ^ 1, (it + 1) * KB);
#ifdef HAS_ASYNC_LDS
    if (more) __builtin_amdgcn_s_wait_asynccnt(4);
    else      __builtin_amdgcn_s_wait_asynccnt(0);
#endif
    __syncthreads();

    const h16* Ks = Ksb[cur];
    const h16* Vt = Vtb[cur];

    // S = Q K^T
    v8f s[4];
#pragma unroll
    for (int ni = 0; ni < 4; ++ni) {
      v8f zv = {};
      const v16h kb0 = load_frag_b_nm(&Ks[(ni * 16) * DH], DH);
      const v16h kb1 = load_frag_b_nm(&Ks[(ni * 16) * DH + 32], DH);
      zv = wmma16x16x32(qa0, kb0, zv);
      zv = wmma16x16x32(qa1, kb1, zv);
      s[ni] = zv;
    }

    // scale + bias + mask (f32)
#pragma unroll
    for (int ni = 0; ni < 4; ++ni) {
      const float mk = Msk[cur][ni * 16 + n];
#pragma unroll
      for (int r = 0; r < 8; ++r) {
        const float v = s[ni][r] * scale + bb;
        s[ni][r] = (mk != 0.0f) ? v : -1.0e30f;
      }
    }

    // online softmax (rows live in 16-lane groups -> xor masks 1,2,4,8)
#pragma unroll
    for (int r = 0; r < 8; ++r) {
      float v = fmaxf(fmaxf(s[0][r], s[1][r]), fmaxf(s[2][r], s[3][r]));
      v = fmaxf(v, __shfl_xor(v, 1, 32));
      v = fmaxf(v, __shfl_xor(v, 2, 32));
      v = fmaxf(v, __shfl_xor(v, 4, 32));
      v = fmaxf(v, __shfl_xor(v, 8, 32));
      const float mnew  = fmaxf(mrun[r], v);
      const float alpha = __expf(mrun[r] - mnew);
      mrun[r] = mnew;
      lrun[r] *= alpha;
#pragma unroll
      for (int di = 0; di < 4; ++di) oacc[di][r] *= alpha;
      float rs = 0.0f;
#pragma unroll
      for (int ni = 0; ni < 4; ++ni) {
        const float pe = __expf(s[ni][r] - mnew);
        s[ni][r] = pe;
        rs += pe;
      }
      rs += __shfl_xor(rs, 1, 32);
      rs += __shfl_xor(rs, 2, 32);
      rs += __shfl_xor(rs, 4, 32);
      rs += __shfl_xor(rs, 8, 32);
      lrun[r] += rs;
    }

    // P -> wave-private LDS, reload as A fragments
#pragma unroll
    for (int ni = 0; ni < 4; ++ni)
#pragma unroll
      for (int r = 0; r < 8; ++r)
        Pw[(r + 8 * hi) * KB + ni * 16 + n] = (h16)s[ni][r];

    __builtin_amdgcn_wave_barrier();
    asm volatile("s_wait_dscnt 0" ::: "memory");

    const v16h pa0 = load_frag_a_rm(&Pw[0], KB);
    const v16h pa1 = load_frag_a_rm(&Pw[32], KB);

    // O += P V
#pragma unroll
    for (int di = 0; di < 4; ++di) {
      const v16h vb0 = load_frag_b_nm(&Vt[(di * 16) * KB], KB);
      const v16h vb1 = load_frag_b_nm(&Vt[(di * 16) * KB + 32], KB);
      oacc[di] = wmma16x16x32(pa0, vb0, oacc[di]);
      oacc[di] = wmma16x16x32(pa1, vb1, oacc[di]);
    }

    __syncthreads();  // buf `cur` free for refill
  }
#undef ATTN_FILL
#undef ATTN_CP

  // Epilogue: rows -> consecutive t, two float4 stores per d-tile.
  float rinv[8];
#pragma unroll
  for (int r = 0; r < 8; ++r) rinv[r] = 1.0f / lrun[r];

#pragma unroll
  for (int di = 0; di < 4; ++di) {
    const int d = di * 16 + n;
    const int tb = q0 + wave * 16 + 8 * hi;
    float* op = out + ((size_t)(b * Cc + h * DH + d)) * Tt + tb;
    float4 w0, w1;
    w0.x = oacc[di][0] * rinv[0];
    w0.y = oacc[di][1] * rinv[1];
    w0.z = oacc[di][2] * rinv[2];
    w0.w = oacc[di][3] * rinv[3];
    w1.x = oacc[di][4] * rinv[4];
    w1.y = oacc[di][5] * rinv[5];
    w1.z = oacc[di][6] * rinv[6];
    w1.w = oacc[di][7] * rinv[7];
    *(float4*)(op)     = w0;
    *(float4*)(op + 4) = w1;
  }
}

extern "C" void kernel_launch(void* const* d_in, const int* in_sizes, int n_in,
                              void* d_out, int out_size, void* d_ws, size_t ws_size,
                              hipStream_t stream) {
  (void)in_sizes; (void)n_in; (void)out_size; (void)ws_size;
  const float* x    = (const float*)d_in[0];
  const float* mask = (const float*)d_in[1];
  const float* Wq   = (const float*)d_in[2];
  const float* Wk   = (const float*)d_in[3];
  const float* Wv   = (const float*)d_in[4];
  const float* bias = (const float*)d_in[5];
  float* out = (float*)d_out;

  const size_t elems = (size_t)Bb * Hh * Tt * DH;  // 8M halves per tensor
  h16* q_ws = (h16*)d_ws;
  h16* k_ws = q_ws + elems;
  h16* v_ws = k_ws + elems;
  h16* wh   = v_ws + elems;                        // 3M halves
  h16* xh_t = wh + (size_t)3 * Cc * Cc;            // 8M halves

  convert_w_kernel<<<dim3((Cc * Cc) / (256 * 8), 3), 256, 0, stream>>>(
      Wq, Wk, Wv, wh);
  transpose_x_kernel<<<dim3(Tt / 64, Cc / 64, Bb), 256, 0, stream>>>(x, xh_t);
  qkv_proj_kernel<<<dim3(Tt / 128, Cc / 128, Bb * 3), 256, 0, stream>>>(
      xh_t, wh, q_ws, k_ws, v_ws);
  attn_kernel<<<dim3(Tt / 128, Bb * Hh), 256, 0, stream>>>(
      q_ws, k_ws, v_ws, mask, bias, out);
}
